// LocalEncoder_91199335563962
// MI455X (gfx1250) — matrix operand order repaired and checked by reading, not compiled, not run on hardware
//
#include <hip/hip_runtime.h>
#include <math.h>

// ---------------- problem constants (from reference setup_inputs) ------------
#define Bsz   4
#define Lseq  4096
#define Dmod  768
#define DFF   3072
#define Kpat  512
#define Hn    12
#define Hd    64
#define BLrows (Bsz * Lseq)     // 16384
#define BKrows (Bsz * Kpat)     // 2048

typedef __attribute__((ext_vector_type(16))) __bf16 v16bf;
typedef __attribute__((ext_vector_type(8)))  float  v8f;
typedef int v4i_gcc __attribute__((vector_size(16)));   // matches builtin signature

union FragU { uint4 u[2]; v16bf v; };

// -------- CDNA5 async memory->LDS path (ASYNCcnt-tracked) --------------------
#if defined(__has_builtin)
#if __has_builtin(__builtin_amdgcn_global_load_async_to_lds_b128)
#define ASYNC_LDS 1
#endif
#endif
#ifndef ASYNC_LDS
#define ASYNC_LDS 0
#endif

#if ASYNC_LDS
__device__ __forceinline__ void async_b128(const void* g, void* l) {
    // generic->AS1: full 64-bit address; generic->AS3: low 32 bits are LDS offset
    __builtin_amdgcn_global_load_async_to_lds_b128(
        (__attribute__((address_space(1))) v4i_gcc*)(uintptr_t)g,
        (__attribute__((address_space(3))) v4i_gcc*)(uintptr_t)l, 0, 0);
}
__device__ __forceinline__ void wait_async0() {
    asm volatile("s_wait_asynccnt 0x0" ::: "memory");
}
#endif

// ---------------- elementwise f32 -> bf16 cast --------------------------------
__global__ __launch_bounds__(256)
void cast_bf16_kernel(const float* __restrict__ src, __bf16* __restrict__ dst, int n) {
    int i = blockIdx.x * blockDim.x + threadIdx.x;
    if (i < n) dst[i] = (__bf16)src[i];
}

// ---------------- embedding gather + bf16 copy -------------------------------
__global__ __launch_bounds__(256)
void embed_kernel(const int* __restrict__ tokens, const float* __restrict__ embW,
                  float* __restrict__ h, __bf16* __restrict__ hb) {
    int row = blockIdx.x;                       // 0..BLrows-1
    int t = tokens[row];
    const float* src = embW + (size_t)t * Dmod;
    size_t base = (size_t)row * Dmod;
    for (int i = threadIdx.x; i < Dmod; i += blockDim.x) {
        float v = src[i];
        h[base + i]  = v;
        hb[base + i] = (__bf16)v;
    }
}

// ---------------- generic bf16 WMMA GEMM: C = A[M,K] * W[N,K]^T --------------
// mode 0: store fp32 C.  mode 1: boundary-predictor epilogue:
//   contrib = silu(acc + b1[n]) * w2[n], row-reduced, atomicAdd into zout[m].
#define BM  64
#define BN  128
#define BKC 64    // K elements per LDS stage (two WMMA k-steps)

__global__ __launch_bounds__(256)
void gemm_wmma_kernel(const __bf16* __restrict__ A, const __bf16* __restrict__ W,
                      float* __restrict__ C, int M, int N, int Kdim, int mode,
                      const float* __restrict__ b1, const float* __restrict__ w2,
                      float* __restrict__ zout) {
    __shared__ __align__(16) __bf16 As[2][BM * BKC];   // 2 x 8 KB
    __shared__ __align__(16) __bf16 Bs[2][BN * BKC];   // 2 x 16 KB
    __shared__ float rowsum[BM];

    const int tid  = threadIdx.x;
    const int bm   = blockIdx.x * BM;
    const int bn   = blockIdx.y * BN;
    const int wave = tid >> 5;
    const int lane = tid & 31;
    const int wm   = (wave >> 2) * 32;   // 0 or 32
    const int wn   = (wave & 3) * 32;    // 0,32,64,96
    const int l15  = lane & 15;
    const int lhi  = lane >> 4;

    // cooperative-load mapping for one stage
    const int ar = tid >> 2, acol = (tid & 3) * 16;   // A: 64x64 -> 2 b128/thread
    const int br = tid >> 1, bcol = (tid & 1) * 32;   // B: 128x64 -> 4 b128/thread
    const __bf16* gA = A + (size_t)(bm + ar) * Kdim + acol;
    const __bf16* gB = W + (size_t)(bn + br) * Kdim + bcol;

    v8f acc[2][2] = {};

    auto issue_stage = [&](int buf, int k0) {
        const __bf16* ga = gA + k0;
        const __bf16* gb = gB + k0;
        __bf16* la = &As[buf][ar * BKC + acol];
        __bf16* lb = &Bs[buf][br * BKC + bcol];
        if (k0 + BKC < Kdim) {   // speculative prefetch of the stage after next
            __builtin_prefetch(ga + BKC, 0, 1);
            __builtin_prefetch(gb + BKC, 0, 1);
        }
#if ASYNC_LDS
        async_b128(ga,      la);
        async_b128(ga + 8,  la + 8);
        async_b128(gb,      lb);
        async_b128(gb + 8,  lb + 8);
        async_b128(gb + 16, lb + 16);
        async_b128(gb + 24, lb + 24);
#else
        *(uint4*)(la)      = *(const uint4*)(ga);
        *(uint4*)(la + 8)  = *(const uint4*)(ga + 8);
        *(uint4*)(lb)      = *(const uint4*)(gb);
        *(uint4*)(lb + 8)  = *(const uint4*)(gb + 8);
        *(uint4*)(lb + 16) = *(const uint4*)(gb + 16);
        *(uint4*)(lb + 24) = *(const uint4*)(gb + 24);
#endif
    };

    const int nstages = Kdim / BKC;
    issue_stage(0, 0);

    for (int s = 0; s < nstages; ++s) {
        const int cur = s & 1;
#if ASYNC_LDS
        wait_async0();           // this wave's stage-s tile landed in LDS
#endif
        __syncthreads();         // ... and everyone else's too
        if (s + 1 < nstages) issue_stage(cur ^ 1, (s + 1) * BKC);

        for (int ks = 0; ks < 2; ++ks) {   // two 16x16x32 k-steps per stage
            FragU afr[2], bfr[2];
            // A fragment (16x32): lanes0-15 K-base 0, lanes16-31 K-base 8;
            // regs 0..3 = K koff..koff+7, regs 4..7 = K koff+16..koff+23.
            for (int mi = 0; mi < 2; ++mi) {
                const __bf16* p = &As[cur][(wm + mi * 16 + l15) * BKC + ks * 32 + lhi * 8];
                afr[mi].u[0] = *(const uint4*)p;
                afr[mi].u[1] = *(const uint4*)(p + 16);
            }
            // B fragment (32x16): lane holds col n=l15, K=lhi*8..+7 and 16+lhi*8..+7 (same as A).
            for (int ni = 0; ni < 2; ++ni) {
                const __bf16* p = &Bs[cur][(wn + ni * 16 + l15) * BKC + ks * 32 + lhi * 8];
                bfr[ni].u[0] = *(const uint4*)p;
                bfr[ni].u[1] = *(const uint4*)(p + 16);
            }
            for (int mi = 0; mi < 2; ++mi)
                for (int ni = 0; ni < 2; ++ni)
                    acc[mi][ni] = __builtin_amdgcn_wmma_f32_16x16x32_bf16(
                        false, afr[mi].v, false, bfr[ni].v,
                        (short)0, acc[mi][ni], false, false);
        }
        __syncthreads();         // all reads of buf `cur` done before it is refilled
    }

    if (mode == 0) {
        // C/D layout: reg r, lanes 0-15 -> M=r, N=lane; lanes 16-31 -> M=r+8.
        for (int mi = 0; mi < 2; ++mi)
            for (int ni = 0; ni < 2; ++ni)
                for (int r = 0; r < 8; ++r) {
                    int row = bm + wm + mi * 16 + r + lhi * 8;
                    int col = bn + wn + ni * 16 + l15;
                    C[(size_t)row * N + col] = acc[mi][ni][r];
                }
    } else {
        if (tid < BM) rowsum[tid] = 0.f;
        __syncthreads();
        for (int mi = 0; mi < 2; ++mi)
            for (int ni = 0; ni < 2; ++ni)
                for (int r = 0; r < 8; ++r) {
                    int lrow = wm + mi * 16 + r + lhi * 8;
                    int col  = bn + wn + ni * 16 + l15;
                    float y = acc[mi][ni][r] + b1[col];
                    float sgl = y / (1.f + __expf(-y));        // SiLU
                    atomicAdd(&rowsum[lrow], sgl * w2[col]);
                }
        __syncthreads();
        if (tid < BM) atomicAdd(&zout[bm + tid], rowsum[tid]);
    }
}

// ---------------- logsigmoid (+b2, boundary at l==0) -------------------------
__global__ __launch_bounds__(256)
void logsig_kernel(const float* __restrict__ z, const float* __restrict__ b2,
                   float* __restrict__ lp) {
    int i = blockIdx.x * blockDim.x + threadIdx.x;
    if (i >= BLrows) return;
    float x = z[i] + b2[0];
    float v = fminf(x, 0.f) - log1pf(__expf(-fabsf(x)));
    if ((i % Lseq) == 0) v = 0.f;
    lp[i] = v;
}

// ---------------- per-batch top-k via LDS bitonic sort -----------------------
__global__ __launch_bounds__(1024)
void topk_kernel(const float* __restrict__ lp, int* __restrict__ pos) {
    __shared__ float sv[Lseq];
    __shared__ int   si[Lseq];
    __shared__ int   sp[Kpat];
    const int b = blockIdx.x, tid = threadIdx.x;

    for (int i = tid; i < Lseq; i += 1024) { sv[i] = lp[b * Lseq + i]; si[i] = i; }
    __syncthreads();

    // sort full row: value descending, index ascending on ties (== lax.top_k)
    for (int k = 2; k <= Lseq; k <<= 1)
        for (int j = k >> 1; j > 0; j >>= 1) {
            for (int i = tid; i < Lseq; i += 1024) {
                int ix = i ^ j;
                if (ix > i) {
                    bool dir = ((i & k) == 0);
                    float va = sv[i], vb = sv[ix];
                    int   ia = si[i], ib = si[ix];
                    bool ooo = (vb > va) || (vb == va && ib < ia);
                    if (ooo == dir) { sv[i] = vb; sv[ix] = va; si[i] = ib; si[ix] = ia; }
                }
            }
            __syncthreads();
        }

    for (int i = tid; i < Kpat; i += 1024) sp[i] = si[i];
    __syncthreads();

    // sort winning indices ascending
    for (int k = 2; k <= Kpat; k <<= 1)
        for (int j = k >> 1; j > 0; j >>= 1) {
            for (int i = tid; i < Kpat; i += 1024) {
                int ix = i ^ j;
                if (ix > i) {
                    bool dir = ((i & k) == 0);
                    int a = sp[i], c = sp[ix];
                    bool ooo = (c < a);
                    if (ooo == dir) { sp[i] = c; sp[ix] = a; }
                }
            }
            __syncthreads();
        }

    for (int i = tid; i < Kpat; i += 1024) pos[b * Kpat + i] = sp[i];
}

// ---------------- RMSNorm (+ optional boundary gather) -> bf16 ---------------
__global__ __launch_bounds__(256)
void rmsnorm_kernel(const float* __restrict__ x, const float* __restrict__ w,
                    __bf16* __restrict__ out, const int* __restrict__ pos,
                    int useGather) {
    __shared__ float red[256];
    const int r = blockIdx.x;
    size_t src;
    if (useGather) { int b = r / Kpat; src = (size_t)(b * Lseq + pos[r]) * Dmod; }
    else           src = (size_t)r * Dmod;
    const float* xr = x + src;

    float ss = 0.f;
    for (int i = threadIdx.x; i < Dmod; i += 256) { float v = xr[i]; ss += v * v; }
    red[threadIdx.x] = ss;
    __syncthreads();
    for (int s = 128; s > 0; s >>= 1) {
        if (threadIdx.x < s) red[threadIdx.x] += red[threadIdx.x + s];
        __syncthreads();
    }
    float scale = rsqrtf(red[0] / (float)Dmod + 1e-5f);
    for (int i = threadIdx.x; i < Dmod; i += 256)
        out[(size_t)r * Dmod + i] = (__bf16)(xr[i] * scale * w[i]);
}

// ---------------- windowed cross-attention (patch ranges are contiguous) -----
__global__ __launch_bounds__(64)
void attn_kernel(const float* __restrict__ q, const float* __restrict__ kmat,
                 const float* __restrict__ vmat, const int* __restrict__ pos,
                 __bf16* __restrict__ outb) {
    __shared__ float sc[Lseq];          // max window < 4096
    __shared__ float red[64];
    const int bq = blockIdx.x;          // 0..BKrows-1
    const int b  = bq / Kpat, kq = bq % Kpat;
    const int tid = threadIdx.x;
    const int start = pos[b * Kpat + kq];
    const int end   = (kq == Kpat - 1) ? Lseq : pos[b * Kpat + kq + 1];
    const int wlen  = end - start;

    for (int h = 0; h < Hn; ++h) {
        const float* qv = q + ((size_t)bq * Dmod + h * Hd);
        // scores
        for (int l = tid; l < wlen; l += 64) {
            const float* kv = kmat + ((size_t)(b * Lseq + start + l) * Dmod + h * Hd);
            float s = 0.f;
            #pragma unroll 8
            for (int d = 0; d < Hd; ++d) s += qv[d] * kv[d];
            sc[l] = s * 0.125f;          // 1/sqrt(64)
        }
        __syncthreads();
        // max
        float m = -3.4e38f;
        for (int l = tid; l < wlen; l += 64) m = fmaxf(m, sc[l]);
        red[tid] = m; __syncthreads();
        for (int s = 32; s > 0; s >>= 1) {
            if (tid < s) red[tid] = fmaxf(red[tid], red[tid + s]);
            __syncthreads();
        }
        m = red[0]; __syncthreads();
        // exp + sum
        float ssum = 0.f;
        for (int l = tid; l < wlen; l += 64) { float e = __expf(sc[l] - m); sc[l] = e; ssum += e; }
        red[tid] = ssum; __syncthreads();
        for (int s = 32; s > 0; s >>= 1) {
            if (tid < s) red[tid] += red[tid + s];
            __syncthreads();
        }
        float inv = 1.f / red[0];
        // weighted sum of V, thread owns output dim d = tid
        float acc = 0.f;
        for (int l = 0; l < wlen; ++l)
            acc += sc[l] * vmat[(size_t)(b * Lseq + start + l) * Dmod + h * Hd + tid];
        outb[(size_t)bq * Dmod + h * Hd + tid] = (__bf16)(acc * inv);
        __syncthreads();
    }
}

// ---------------- host-side orchestration ------------------------------------
extern "C" void kernel_launch(void* const* d_in, const int* in_sizes, int n_in,
                              void* d_out, int out_size, void* d_ws, size_t ws_size,
                              hipStream_t stream) {
    const int*   tokens   = (const int*)  d_in[0];
    const float* embW     = (const float*)d_in[1];
    const float* bp_w1    = (const float*)d_in[2];
    const float* bp_b1    = (const float*)d_in[3];
    const float* bp_w2    = (const float*)d_in[4];
    const float* bp_b2    = (const float*)d_in[5];
    const float* wq       = (const float*)d_in[6];
    const float* wk       = (const float*)d_in[7];
    const float* wv       = (const float*)d_in[8];
    const float* wo       = (const float*)d_in[9];
    const float* qnorm_w  = (const float*)d_in[10];
    const float* kvnorm_w = (const float*)d_in[11];
    (void)in_sizes; (void)n_in; (void)out_size; (void)ws_size;

    char* w = (char*)d_ws;
    size_t off = 0;
    auto alloc = [&](size_t bytes) -> void* {
        void* p = w + off;
        off = (off + bytes + 255) & ~(size_t)255;
        return p;
    };

    float*  h_f32  = (float*) alloc((size_t)BLrows * Dmod * 4);
    __bf16* h_bf   = (__bf16*)alloc((size_t)BLrows * Dmod * 2);
    __bf16* kvn_bf = (__bf16*)alloc((size_t)BLrows * Dmod * 2);
    float*  z      = (float*) alloc((size_t)BLrows * 4);
    float*  logp   = (float*) alloc((size_t)BLrows * 4);
    int*    pos    = (int*)   alloc((size_t)Bsz * Kpat * 4);
    __bf16* qn_bf  = (__bf16*)alloc((size_t)BKrows * Dmod * 2);
    float*  qf     = (float*) alloc((size_t)BKrows * Dmod * 4);
    float*  kf     = (float*) alloc((size_t)BLrows * Dmod * 4);
    float*  vf     = (float*) alloc((size_t)BLrows * Dmod * 4);
    __bf16* aout   = (__bf16*)alloc((size_t)BKrows * Dmod * 2);
    __bf16* w1_bf  = (__bf16*)alloc((size_t)DFF * Dmod * 2);
    __bf16* wq_bf  = (__bf16*)alloc((size_t)Dmod * Dmod * 2);
    __bf16* wk_bf  = (__bf16*)alloc((size_t)Dmod * Dmod * 2);
    __bf16* wv_bf  = (__bf16*)alloc((size_t)Dmod * Dmod * 2);
    __bf16* wo_bf  = (__bf16*)alloc((size_t)Dmod * Dmod * 2);

    (void)hipMemsetAsync(z, 0, (size_t)BLrows * 4, stream);

    // weight casts
    {
        int n1 = DFF * Dmod, n2 = Dmod * Dmod;
        cast_bf16_kernel<<<(n1 + 255) / 256, 256, 0, stream>>>(bp_w1, w1_bf, n1);
        cast_bf16_kernel<<<(n2 + 255) / 256, 256, 0, stream>>>(wq, wq_bf, n2);
        cast_bf16_kernel<<<(n2 + 255) / 256, 256, 0, stream>>>(wk, wk_bf, n2);
        cast_bf16_kernel<<<(n2 + 255) / 256, 256, 0, stream>>>(wv, wv_bf, n2);
        cast_bf16_kernel<<<(n2 + 255) / 256, 256, 0, stream>>>(wo, wo_bf, n2);
    }

    // 1) embedding
    embed_kernel<<<BLrows, 256, 0, stream>>>(tokens, embW, h_f32, h_bf);

    // 2) boundary predictor GEMM with fused SiLU * w2 row-reduction
    gemm_wmma_kernel<<<dim3(BLrows / BM, DFF / BN), 256, 0, stream>>>(
        h_bf, w1_bf, nullptr, BLrows, DFF, Dmod, 1, bp_b1, bp_w2, z);

    // 2b) logsigmoid
    logsig_kernel<<<(BLrows + 255) / 256, 256, 0, stream>>>(z, bp_b2, logp);

    // 3) top-k boundaries per batch row
    topk_kernel<<<Bsz, 1024, 0, stream>>>(logp, pos);

    // 4/5) RMSNorms (query path gathers boundary rows)
    rmsnorm_kernel<<<BLrows, 256, 0, stream>>>(h_f32, kvnorm_w, kvn_bf, nullptr, 0);
    rmsnorm_kernel<<<BKrows, 256, 0, stream>>>(h_f32, qnorm_w, qn_bf, pos, 1);

    // projections
    gemm_wmma_kernel<<<dim3(BKrows / BM, Dmod / BN), 256, 0, stream>>>(
        qn_bf, wq_bf, qf, BKrows, Dmod, Dmod, 0, nullptr, nullptr, nullptr);
    gemm_wmma_kernel<<<dim3(BLrows / BM, Dmod / BN), 256, 0, stream>>>(
        kvn_bf, wk_bf, kf, BLrows, Dmod, Dmod, 0, nullptr, nullptr, nullptr);
    gemm_wmma_kernel<<<dim3(BLrows / BM, Dmod / BN), 256, 0, stream>>>(
        kvn_bf, wv_bf, vf, BLrows, Dmod, Dmod, 0, nullptr, nullptr, nullptr);

    // windowed attention over contiguous patch ranges
    attn_kernel<<<BKrows, 64, 0, stream>>>(qf, kf, vf, pos, aout);

    // output projection straight into d_out (fp32)
    gemm_wmma_kernel<<<dim3(BKrows / BM, Dmod / BN), 256, 0, stream>>>(
        aout, wo_bf, (float*)d_out, BKrows, Dmod, Dmod, 0, nullptr, nullptr, nullptr);
}
